// AutoEncoder_77841987273292
// MI455X (gfx1250) — matrix operand order repaired and checked by reading, not compiled, not run on hardware
//
#include <hip/hip_runtime.h>

typedef __attribute__((ext_vector_type(16))) _Float16 v16h;
typedef __attribute__((ext_vector_type(8)))  _Float16 v8h;
typedef __attribute__((ext_vector_type(8)))  float    v8f;

#define BATCH 8
#define NPTS  2048
#define KNBR  17

__device__ __forceinline__ float gelu_f(float x) {
  return 0.5f * x * (1.0f + erff(x * 0.7071067811865476f));
}

// =====================================================================
// Batched GEMM with WMMA f16 (f32 accumulate), double-buffered LDS:
//   C[b] = act( A[b](M,K) @ W(N,K)^T + bias [+ R[b]] )
// Requirements (all call sites satisfy): M%64==0, N%64==0, K%32==0.
// Block = 256 thr (8 waves, 4x2), tile 64x64, K-step 32, 2 WMMA/wave/step.
// act: 0=none, 1=relu, 2=gelu.
// =====================================================================
__global__ __launch_bounds__(256) void k_gemm(
    const float* __restrict__ A, const float* __restrict__ W,
    const float* __restrict__ bias, const float* __restrict__ R,
    float* __restrict__ C, int M, int N, int K, int act)
{
  __shared__ __align__(16) _Float16 a_lds[2][64][40];
  __shared__ __align__(16) _Float16 w_lds[2][64][40];
  const int b = blockIdx.z;
  const float* Ab = A + (size_t)b * M * K;
  float* Cb = C + (size_t)b * M * N;
  const float* Rb = R ? (R + (size_t)b * M * N) : nullptr;
  const int tid  = threadIdx.x;
  const int wave = tid >> 5, lane = tid & 31;
  const int wm = wave & 3;        // 4 M sub-tiles of 16
  const int wn = wave >> 2;       // 2 N sub-tiles of 32
  const int m0 = blockIdx.y * 64, n0 = blockIdx.x * 64;
  // cooperative staging: 64 rows x 32 k halves, 8 halves (2x float4) per thread
  const int row = tid >> 2, col = (tid & 3) * 8;
  const float* arow_p = Ab + (size_t)(m0 + row) * K + col;
  const float* wrow_p = W  + (size_t)(n0 + row) * K + col;

  auto stage = [&](int k0, int buf) {
    float4 a0 = *(const float4*)(arow_p + k0);
    float4 a1 = *(const float4*)(arow_p + k0 + 4);
    float4 w0 = *(const float4*)(wrow_p + k0);
    float4 w1 = *(const float4*)(wrow_p + k0 + 4);
    _Float16* ap = &a_lds[buf][row][col];
    ap[0] = (_Float16)a0.x; ap[1] = (_Float16)a0.y; ap[2] = (_Float16)a0.z; ap[3] = (_Float16)a0.w;
    ap[4] = (_Float16)a1.x; ap[5] = (_Float16)a1.y; ap[6] = (_Float16)a1.z; ap[7] = (_Float16)a1.w;
    _Float16* wp = &w_lds[buf][row][col];
    wp[0] = (_Float16)w0.x; wp[1] = (_Float16)w0.y; wp[2] = (_Float16)w0.z; wp[3] = (_Float16)w0.w;
    wp[4] = (_Float16)w1.x; wp[5] = (_Float16)w1.y; wp[6] = (_Float16)w1.z; wp[7] = (_Float16)w1.w;
    // deep prefetch of the k+2 block (global_prefetch_b8 path)
    if (k0 + 64 < K) {
      __builtin_prefetch(arow_p + k0 + 64, 0, 0);
      __builtin_prefetch(wrow_p + k0 + 64, 0, 0);
    }
  };

  v8f acc0 = {}, acc1 = {};
  const int nk = K >> 5;
  stage(0, 0);
  for (int ks = 0; ks < nk; ks++) {
    const int buf = ks & 1;
    __syncthreads();                       // stage(buf) visible; prev reads of buf^1 done
    if (ks + 1 < nk) stage((ks + 1) << 5, buf ^ 1);
    // A fragment (16x32 MxK): lanes<16 hold K 0-7,16-23 ; lanes>=16 hold K 8-15,24-31
    const int ml = wm * 16 + (lane & 15);
    const int hk = lane >> 4;
    v8h alo = *(const v8h*)(&a_lds[buf][ml][hk * 8]);
    v8h ahi = *(const v8h*)(&a_lds[buf][ml][16 + hk * 8]);
    v16h afrag;
#pragma unroll
    for (int i = 0; i < 8; i++) { afrag[i] = alo[i]; afrag[i + 8] = ahi[i]; }
    // B fragments (KxN = W^T): lane holds column n=lane%16, K hk*8..+7 and 16+hk*8..+7
    const int nl0 = wn * 32 + (lane & 15);
    v8h b0lo = *(const v8h*)(&w_lds[buf][nl0][hk * 8]);
    v8h b0hi = *(const v8h*)(&w_lds[buf][nl0][16 + hk * 8]);
    v16h bfrag0;
#pragma unroll
    for (int i = 0; i < 8; i++) { bfrag0[i] = b0lo[i]; bfrag0[i + 8] = b0hi[i]; }
    const int nl1 = nl0 + 16;
    v8h b1lo = *(const v8h*)(&w_lds[buf][nl1][hk * 8]);
    v8h b1hi = *(const v8h*)(&w_lds[buf][nl1][16 + hk * 8]);
    v16h bfrag1;
#pragma unroll
    for (int i = 0; i < 8; i++) { bfrag1[i] = b1lo[i]; bfrag1[i + 8] = b1hi[i]; }
    acc0 = __builtin_amdgcn_wmma_f32_16x16x32_f16(
        false, afrag, false, bfrag0, (short)0, acc0, false, false);
    acc1 = __builtin_amdgcn_wmma_f32_16x16x32_f16(
        false, afrag, false, bfrag1, (short)0, acc1, false, false);
  }
  // Epilogue: C/D layout: VGPR r -> M = r + (lane>=16)*8, N = lane%16
  const int mb  = m0 + wm * 16 + (lane >> 4) * 8;
  const int ng0 = n0 + wn * 32 + (lane & 15);
  const int ng1 = ng0 + 16;
  float bv0 = bias ? bias[ng0] : 0.f;
  float bv1 = bias ? bias[ng1] : 0.f;
#pragma unroll
  for (int r = 0; r < 8; r++) {
    const int mg = mb + r;
    float v0 = acc0[r] + bv0;
    float v1 = acc1[r] + bv1;
    if (Rb) { v0 += Rb[(size_t)mg * N + ng0]; v1 += Rb[(size_t)mg * N + ng1]; }
    if (act == 1) { v0 = fmaxf(v0, 0.f); v1 = fmaxf(v1, 0.f); }
    else if (act == 2) { v0 = gelu_f(v0); v1 = gelu_f(v1); }
    Cb[(size_t)mg * N + ng0] = v0;
    Cb[(size_t)mg * N + ng1] = v1;
  }
}

// ================= LayerNorm over channel (one wave per token) =================
__global__ __launch_bounds__(32) void k_ln(float* __restrict__ X,
    const float* __restrict__ g, const float* __restrict__ bta, int C)
{
  float* xp = X + (size_t)blockIdx.x * C;
  const int lane = threadIdx.x;
  float vals[16];
  int cnt = 0; float s = 0.f;
  for (int c = lane; c < C; c += 32) { vals[cnt] = xp[c]; s += vals[cnt]; cnt++; }
  for (int o = 16; o > 0; o >>= 1) s += __shfl_xor(s, o, 32);
  float mean = s / (float)C;
  float vs = 0.f;
  for (int i = 0; i < cnt; i++) { float d = vals[i] - mean; vs += d * d; }
  for (int o = 16; o > 0; o >>= 1) vs += __shfl_xor(vs, o, 32);
  float inv = rsqrtf(vs / (float)C + 1e-5f);
  cnt = 0;
  for (int c = lane; c < C; c += 32) { xp[c] = (vals[cnt] - mean) * inv * g[c] + bta[c]; cnt++; }
}

// ===== Flash attention: one wave per (query, head, batch), online softmax =====
__global__ __launch_bounds__(32) void k_attn(const float* __restrict__ Q,
    const float* __restrict__ Km, const float* __restrict__ V,
    float* __restrict__ O, int N1, int N2, int C, int H)
{
  const int qi = blockIdx.x, h = blockIdx.y, b = blockIdx.z;
  const int hd = C / H;
  const float scale = rsqrtf((float)hd);
  const int lane = threadIdx.x;
  const int per = (hd + 31) / 32;
  const float* qp = Q + ((size_t)(b * N1 + qi)) * C + h * hd;
  float qr[4] = {0.f, 0.f, 0.f, 0.f};
  for (int i = 0; i < per; i++) { int d = lane + i * 32; if (d < hd) qr[i] = qp[d]; }
  const float* Kb = Km + ((size_t)b * N2) * C + h * hd;
  const float* Vb = V  + ((size_t)b * N2) * C + h * hd;
  float mx = -3.4e38f, S = 0.f;
  float oa[4] = {0.f, 0.f, 0.f, 0.f};
  for (int j = 0; j < N2; j++) {
    const float* kp = Kb + (size_t)j * C;
    float p = 0.f;
    for (int i = 0; i < per; i++) { int d = lane + i * 32; if (d < hd) p += qr[i] * kp[d]; }
    for (int o = 16; o > 0; o >>= 1) p += __shfl_xor(p, o, 32);
    float s = p * scale;
    float mn = fmaxf(mx, s);
    float al = expf(mx - mn);
    float pw = expf(s - mn);
    S = S * al + pw;
    const float* vp = Vb + (size_t)j * C;
    for (int i = 0; i < per; i++) { int d = lane + i * 32; if (d < hd) oa[i] = oa[i] * al + pw * vp[d]; }
    mx = mn;
  }
  float inv = 1.f / S;
  float* op = O + ((size_t)(b * N1 + qi)) * C + h * hd;
  for (int i = 0; i < per; i++) { int d = lane + i * 32; if (d < hd) op[d] = oa[i] * inv; }
}

// ========== KNN top-17 (smallest d, first-index ties), one thread/query ==========
__global__ void k_knn(const float* __restrict__ F, int N, int C,
                      int* __restrict__ out, int BN)
{
  int t = blockIdx.x * blockDim.x + threadIdx.x;
  if (t >= BN) return;
  int b = t / N, n = t % N;
  const float* base = F + (size_t)b * N * C;
  const float* q = base + (size_t)n * C;
  float qr[16]; float qn = 0.f;
  for (int c = 0; c < C; c++) { qr[c] = q[c]; qn += qr[c] * qr[c]; }
  float bd[KNBR]; int bi[KNBR];
  for (int j = 0; j < KNBR; j++) { bd[j] = 3.4e38f; bi[j] = 0; }
  for (int m = 0; m < N; m++) {
    const float* p = base + (size_t)m * C;
    float dot = 0.f, pn = 0.f;
    for (int c = 0; c < C; c++) { float v = p[c]; dot += qr[c] * v; pn += v * v; }
    float d = qn - 2.f * dot + pn;
    if (d < bd[KNBR - 1]) {
      int pos = KNBR - 1;
      while (pos > 0 && bd[pos - 1] > d) { bd[pos] = bd[pos - 1]; bi[pos] = bi[pos - 1]; pos--; }
      bd[pos] = d; bi[pos] = m;
    }
  }
  int* op = out + (size_t)t * KNBR;
  for (int j = 0; j < KNBR; j++) op[j] = bi[j];
}

// ===== geo grouping: conv(6->32)+gelu, max over K neighbors, writes x0[:, :32] =====
__global__ void k_geo_max(const float* __restrict__ X, const int* __restrict__ idx,
    const float* __restrict__ w, const float* __restrict__ bias,
    float* __restrict__ x0, int N, int total)
{
  int t = blockIdx.x * blockDim.x + threadIdx.x;
  if (t >= total) return;
  int o = t & 31; int n = (t >> 5) % N; int b = t / (32 * N);
  const float* Xb = X + (size_t)b * N * 3;
  const float* rp = Xb + (size_t)n * 3;
  float r0 = rp[0], r1 = rp[1], r2 = rp[2];
  const int* ip = idx + ((size_t)b * N + n) * KNBR;
  const float* wo = w + o * 6;
  float base = bias[o] + r0 * wo[0] + r1 * wo[1] + r2 * wo[2]
                       - (r0 * wo[3] + r1 * wo[4] + r2 * wo[5]);
  float mx = -3.4e38f;
  for (int k = 0; k < KNBR; k++) {
    const float* pp = Xb + (size_t)ip[k] * 3;
    float s = base + pp[0] * wo[3] + pp[1] * wo[4] + pp[2] * wo[5];
    mx = fmaxf(mx, gelu_f(s));
  }
  x0[((size_t)b * N + n) * 64 + o] = mx;
}

// ===== feature grouping: conv(32->32)+gelu, max over K, writes x0[:, 32:64] =====
__global__ void k_feat_max(const float* __restrict__ F, const int* __restrict__ idx,
    const float* __restrict__ w, const float* __restrict__ bias,
    float* __restrict__ x0, int N, int total)
{
  int t = blockIdx.x * blockDim.x + threadIdx.x;
  if (t >= total) return;
  int o = t & 31; int n = (t >> 5) % N; int b = t / (32 * N);
  const float* Fb = F + (size_t)b * N * 16;
  const float* rp = Fb + (size_t)n * 16;
  const int* ip = idx + ((size_t)b * N + n) * KNBR;
  const float* wo = w + o * 32;
  float base = bias[o];
  for (int c = 0; c < 16; c++) base += rp[c] * (wo[c] - wo[16 + c]);
  float mx = -3.4e38f;
  for (int k = 0; k < KNBR; k++) {
    const float* pp = Fb + (size_t)ip[k] * 16;
    float s = base;
    for (int c = 0; c < 16; c++) s += pp[c] * wo[16 + c];
    mx = fmaxf(mx, gelu_f(s));
  }
  x0[((size_t)b * N + n) * 64 + 32 + o] = mx;
}

// ========== Farthest point sampling: one block per batch, LDS dist array ==========
__global__ __launch_bounds__(256) void k_fps(const float* __restrict__ P,
    int N, int npoint, int* __restrict__ idx)
{
  __shared__ float dist[NPTS];
  __shared__ float rv[256];
  __shared__ int   ri[256];
  const int b = blockIdx.x, tid = threadIdx.x;
  const float* Pb = P + (size_t)b * N * 3;
  for (int i = tid; i < N; i += 256) dist[i] = 3.4e38f;
  __syncthreads();
  int last = 0;
  for (int t = 0; t < npoint; t++) {
    if (tid == 0) idx[b * npoint + t] = last;
    float lx = Pb[last * 3], ly = Pb[last * 3 + 1], lz = Pb[last * 3 + 2];
    float bd = -1.f; int bi = 0x7fffffff;
    for (int i = tid; i < N; i += 256) {
      float dx = Pb[i * 3] - lx, dy = Pb[i * 3 + 1] - ly, dz = Pb[i * 3 + 2] - lz;
      float d = dx * dx + dy * dy + dz * dz;
      float nd = fminf(dist[i], d); dist[i] = nd;
      if (nd > bd) { bd = nd; bi = i; }
    }
    rv[tid] = bd; ri[tid] = bi;
    __syncthreads();
    for (int s = 128; s > 0; s >>= 1) {
      if (tid < s) {
        if (rv[tid + s] > rv[tid] ||
            (rv[tid + s] == rv[tid] && ri[tid + s] < ri[tid])) {
          rv[tid] = rv[tid + s]; ri[tid] = ri[tid + s];
        }
      }
      __syncthreads();
    }
    last = ri[0];
    __syncthreads();
  }
}

// ================================ small kernels ================================
__global__ void k_conv1(const float* __restrict__ X, const float* __restrict__ w,
                        const float* __restrict__ bias, float* __restrict__ out, int total)
{
  int t = blockIdx.x * blockDim.x + threadIdx.x;
  if (t >= total) return;
  int o = t & 15; int n = (t >> 4) % NPTS; int b = t / (16 * NPTS);
  const float* xp = X + ((size_t)b * NPTS + n) * 3;
  float s = bias[o] + xp[0] * w[o * 3] + xp[1] * w[o * 3 + 1] + xp[2] * w[o * 3 + 2];
  out[((size_t)b * NPTS + n) * 16 + o] = gelu_f(s);
}

__global__ void k_gather(const float* __restrict__ in, const int* __restrict__ idx,
                         float* __restrict__ out, int N, int M, int C, int total)
{
  int t = blockIdx.x * blockDim.x + threadIdx.x;
  if (t >= total) return;
  int c = t % C; int i = (t / C) % M; int b = t / (C * M);
  out[t] = in[((size_t)b * N + idx[b * M + i]) * C + c];
}

__global__ void k_concat(const float* __restrict__ A, const float* __restrict__ Bp,
                         float* __restrict__ out, int N, int Ca, int Cb, int total)
{
  int t = blockIdx.x * blockDim.x + threadIdx.x;
  if (t >= total) return;
  int Ct = Ca + Cb;
  int c = t % Ct; int n = (t / Ct) % N; int b = t / (Ct * N);
  out[t] = (c < Ca) ? A[((size_t)b * N + n) * Ca + c]
                    : Bp[((size_t)b * N + n) * Cb + (c - Ca)];
}

__global__ void k_maxpool(const float* __restrict__ X, float* __restrict__ out,
                          int T, int C, int total)
{
  int t = blockIdx.x * blockDim.x + threadIdx.x;
  if (t >= total) return;
  int c = t % C; int b = t / C;
  float m = -3.4e38f;
  for (int tok = 0; tok < T; tok++) m = fmaxf(m, X[((size_t)b * T + tok) * C + c]);
  out[t] = m;
}

// seed[b][t][o] = sum_i fg[b][i] * ps_w[i][o][t] + ps_b[o]; each ps_w elem read once
__global__ __launch_bounds__(256) void k_seed(const float* __restrict__ PW,
    const float* __restrict__ pb, const float* __restrict__ fg, float* __restrict__ seed)
{
  __shared__ float sfg[BATCH * 512];
  int tid = threadIdx.x;
  for (int i = tid; i < BATCH * 512; i += 256) sfg[i] = fg[i];
  __syncthreads();
  int t = blockIdx.x * 256 + tid;   // 0..511
  int o = blockIdx.y;               // 0..127
  float acc[BATCH];
#pragma unroll
  for (int bb = 0; bb < BATCH; bb++) acc[bb] = 0.f;
  const float* wp = PW + (size_t)o * 512 + t;
  for (int i = 0; i < 512; i++) {
    float w = wp[(size_t)i * 65536];
#pragma unroll
    for (int bb = 0; bb < BATCH; bb++) acc[bb] += sfg[bb * 512 + i] * w;
  }
  float bv = pb[o];
#pragma unroll
  for (int bb = 0; bb < BATCH; bb++)
    seed[(((size_t)bb * 512) + t) * 128 + o] = acc[bb] + bv;
}

// cat640[b][t][0:128] = A, [128:640] = feat_g broadcast
__global__ void k_cat_fr(const float* __restrict__ A, const float* __restrict__ fg,
                         float* __restrict__ out, int total)
{
  int t = blockIdx.x * blockDim.x + threadIdx.x;
  if (t >= total) return;
  int c = t % 640; int tok = (t / 640) % 512; int b = t / (640 * 512);
  out[t] = (c < 128) ? A[((size_t)b * 512 + tok) * 128 + c] : fg[b * 512 + (c - 128)];
}

// final 64->3 head, writes comp into d_out tail
__global__ void k_head(const float* __restrict__ H, const float* __restrict__ w,
                       const float* __restrict__ bias, float* __restrict__ out, int total)
{
  int t = blockIdx.x * blockDim.x + threadIdx.x;
  if (t >= total) return;
  int o = t % 3; int tok = (t / 3) % 512; int b = t / (3 * 512);
  const float* hp = H + ((size_t)b * 512 + tok) * 64;
  const float* wo = w + o * 64;
  float s = bias[o];
  for (int c = 0; c < 64; c++) s += hp[c] * wo[c];
  out[t] = s;
}

// =====================================================================
//                              host side
// =====================================================================
extern "C" void kernel_launch(void* const* d_in, const int* in_sizes, int n_in,
                              void* d_out, int out_size, void* d_ws, size_t ws_size,
                              hipStream_t stream) {
  (void)in_sizes; (void)n_in; (void)out_size; (void)ws_size;
  // input index map (setup_inputs dict insertion order, nested dicts in order)
  enum { IN_X = 0, IN_C1W, IN_C1B, IN_GW, IN_GB, IN_FW, IN_FB,
         SA1 = 7, SA1_1 = 25, SA2 = 43, SA2_1 = 61, SA3 = 79, SA3_1 = 97,
         IN_PSW = 115, IN_PSB = 116, M1 = 117, M2 = 123, M3 = 129,
         IN_4AW = 135, IN_4AB = 136, IN_4BW = 137, IN_4BB = 138 };
  auto F = [&](int i) { return (const float*)d_in[i]; };
  const float* X = F(IN_X);
  float* out = (float*)d_out;

  char* ws = (char*)d_ws;
  size_t cur = 0;
  auto allocf = [&](size_t elems) {
    float* p = (float*)(ws + cur);
    cur = (cur + elems * sizeof(float) + 255) & ~(size_t)255;
    return p;
  };
  auto alloci = [&](size_t elems) {
    int* p = (int*)(ws + cur);
    cur = (cur + elems * sizeof(int) + 255) & ~(size_t)255;
    return p;
  };

  float* feat  = allocf((size_t)BATCH * NPTS * 16);
  int*   idxg  = alloci((size_t)BATCH * NPTS * KNBR);
  int*   idxf  = alloci((size_t)BATCH * NPTS * KNBR);
  float* x0    = allocf((size_t)BATCH * NPTS * 64);
  int*   idx0  = alloci((size_t)BATCH * 512);
  int*   idx1  = alloci((size_t)BATCH * 256);
  int*   idx2  = alloci((size_t)BATCH * 128);
  float* pts512= allocf((size_t)BATCH * 512 * 3);
  float* pts256= allocf((size_t)BATCH * 256 * 3);
  float* xg0   = allocf((size_t)BATCH * 512 * 64);
  float* s1b   = allocf((size_t)BATCH * 512 * 128);   // max N1*C = 65536
  float* s2b   = allocf((size_t)BATCH * NPTS * 64);   // max N2*C = 131072
  float* qb    = allocf((size_t)BATCH * 512 * 128);
  float* kb    = allocf((size_t)BATCH * NPTS * 64);
  float* vb    = allocf((size_t)BATCH * NPTS * 64);
  float* ob    = allocf((size_t)BATCH * 512 * 128);
  float* ffb   = allocf((size_t)BATCH * 512 * 1024);
  float* x1a   = allocf((size_t)BATCH * 512 * 64);
  float* x1cat = allocf((size_t)BATCH * 512 * 128);
  float* x1f   = allocf((size_t)BATCH * 512 * 128);
  float* xg1   = allocf((size_t)BATCH * 256 * 128);
  float* x2a   = allocf((size_t)BATCH * 256 * 128);
  float* x2cat = allocf((size_t)BATCH * 256 * 256);
  float* x2f   = allocf((size_t)BATCH * 256 * 256);
  float* xg2   = allocf((size_t)BATCH * 128 * 256);
  float* x3a   = allocf((size_t)BATCH * 128 * 256);
  float* x3cat = allocf((size_t)BATCH * 128 * 512);
  float* x3f   = allocf((size_t)BATCH * 128 * 512);
  float* seed  = allocf((size_t)BATCH * 512 * 128);
  float* cat640= allocf((size_t)BATCH * 512 * 640);
  float* hbuf  = allocf((size_t)BATCH * 512 * 128);
  float* scb   = allocf((size_t)BATCH * 512 * 128);
  float* h1    = allocf((size_t)BATCH * 512 * 128);
  float* h2    = allocf((size_t)BATCH * 512 * 128);
  float* h3    = allocf((size_t)BATCH * 512 * 128);
  float* t64   = allocf((size_t)BATCH * 512 * 64);

  auto blk1 = [](int n) { return dim3((unsigned)((n + 255) / 256)); };

  auto gemm = [&](const float* A, const float* W, const float* bias,
                  const float* R, float* C, int M, int N, int K, int act) {
    dim3 g((N + 63) / 64, (M + 63) / 64, BATCH);
    k_gemm<<<g, dim3(256), 0, stream>>>(A, W, bias, R, C, M, N, K, act);
  };
  auto ln = [&](float* Xp, const float* g, const float* b, int Ntok, int C) {
    k_ln<<<dim3((unsigned)(BATCH * Ntok)), dim3(32), 0, stream>>>(Xp, g, b, C);
  };
  // p offsets in each sa dict: proj_w=0 proj_b=1 ln13_g=2 ln13_b=3 ln12_g=4 ln12_b=5
  // wq=6 bq=7 wk=8 bk=9 wv=10 bv=11 wo=12 bo=13 w1=14 b1=15 w2=16 b2=17
  auto cross = [&](int pb, const float* src1, int N1, const float* src2, int N2,
                   int C, float* dst) {
    gemm(src1, F(pb + 0), F(pb + 1), nullptr, s1b, N1, C, C, 0);
    ln(s1b, F(pb + 2), F(pb + 3), N1, C);
    const float* s2 = s1b;
    if (src2 != src1) {
      gemm(src2, F(pb + 0), F(pb + 1), nullptr, s2b, N2, C, C, 0);
      ln(s2b, F(pb + 2), F(pb + 3), N2, C);
      s2 = s2b;
    }
    gemm(s1b, F(pb + 6), F(pb + 7), nullptr, qb, N1, C, C, 0);
    gemm(s2,  F(pb + 8), F(pb + 9), nullptr, kb, N2, C, C, 0);
    gemm(s2,  F(pb + 10), F(pb + 11), nullptr, vb, N2, C, C, 0);
    k_attn<<<dim3((unsigned)N1, 4, BATCH), dim3(32), 0, stream>>>(qb, kb, vb, ob, N1, N2, C, 4);
    gemm(ob, F(pb + 12), F(pb + 13), s1b, s1b, N1, C, C, 0);  // residual
    ln(s1b, F(pb + 4), F(pb + 5), N1, C);
    gemm(s1b, F(pb + 14), F(pb + 15), nullptr, ffb, N1, 1024, C, 2);
    gemm(ffb, F(pb + 16), F(pb + 17), s1b, dst, N1, C, 1024, 0);  // residual
  };

  // ---- encoder front-end ----
  k_conv1<<<blk1(BATCH * NPTS * 16), 256, 0, stream>>>(X, F(IN_C1W), F(IN_C1B), feat, BATCH * NPTS * 16);
  k_knn<<<blk1(BATCH * NPTS), 256, 0, stream>>>(X, NPTS, 3, idxg, BATCH * NPTS);
  k_knn<<<blk1(BATCH * NPTS), 256, 0, stream>>>(feat, NPTS, 16, idxf, BATCH * NPTS);
  k_geo_max<<<blk1(BATCH * NPTS * 32), 256, 0, stream>>>(X, idxg, F(IN_GW), F(IN_GB), x0, NPTS, BATCH * NPTS * 32);
  k_feat_max<<<blk1(BATCH * NPTS * 32), 256, 0, stream>>>(feat, idxf, F(IN_FW), F(IN_FB), x0, NPTS, BATCH * NPTS * 32);
  k_fps<<<dim3(BATCH), dim3(256), 0, stream>>>(X, NPTS, 512, idx0);
  k_gather<<<blk1(BATCH * 512 * 64), 256, 0, stream>>>(x0, idx0, xg0, NPTS, 512, 64, BATCH * 512 * 64);
  k_gather<<<blk1(BATCH * 512 * 3), 256, 0, stream>>>(X, idx0, pts512, NPTS, 512, 3, BATCH * 512 * 3);

  // ---- transformer stack ----
  cross(SA1, xg0, 512, x0, NPTS, 64, x1a);
  k_concat<<<blk1(BATCH * 512 * 128), 256, 0, stream>>>(xg0, x1a, x1cat, 512, 64, 64, BATCH * 512 * 128);
  cross(SA1_1, x1cat, 512, x1cat, 512, 128, x1f);
  k_fps<<<dim3(BATCH), dim3(256), 0, stream>>>(pts512, 512, 256, idx1);
  k_gather<<<blk1(BATCH * 256 * 128), 256, 0, stream>>>(x1f, idx1, xg1, 512, 256, 128, BATCH * 256 * 128);
  k_gather<<<blk1(BATCH * 256 * 3), 256, 0, stream>>>(pts512, idx1, pts256, 512, 256, 3, BATCH * 256 * 3);
  cross(SA2, xg1, 256, x1f, 512, 128, x2a);
  k_concat<<<blk1(BATCH * 256 * 256), 256, 0, stream>>>(xg1, x2a, x2cat, 256, 128, 128, BATCH * 256 * 256);
  cross(SA2_1, x2cat, 256, x2cat, 256, 256, x2f);
  k_fps<<<dim3(BATCH), dim3(256), 0, stream>>>(pts256, 256, 128, idx2);
  k_gather<<<blk1(BATCH * 128 * 256), 256, 0, stream>>>(x2f, idx2, xg2, 256, 128, 256, BATCH * 128 * 256);
  cross(SA3, xg2, 128, x2f, 256, 256, x3a);
  k_concat<<<blk1(BATCH * 128 * 512), 256, 0, stream>>>(xg2, x3a, x3cat, 128, 256, 256, BATCH * 128 * 512);
  cross(SA3_1, x3cat, 128, x3cat, 128, 512, x3f);

  // ---- global feature (output 1) ----
  k_maxpool<<<blk1(BATCH * 512), 256, 0, stream>>>(x3f, out, 128, 512, BATCH * 512);

  // ---- decoder ----
  k_seed<<<dim3(2, 128), dim3(256), 0, stream>>>(F(IN_PSW), F(IN_PSB), out, seed);
  k_cat_fr<<<blk1(BATCH * 512 * 640), 256, 0, stream>>>(seed, out, cat640, BATCH * 512 * 640);
  // mlp1: w1,b1,w2,b2,ws,bs
  gemm(cat640, F(M1 + 0), F(M1 + 1), nullptr, hbuf, 512, 128, 640, 1);
  gemm(cat640, F(M1 + 4), F(M1 + 5), nullptr, scb, 512, 128, 640, 0);
  gemm(hbuf, F(M1 + 2), F(M1 + 3), scb, h1, 512, 128, 128, 0);
  // mlp2
  gemm(h1, F(M2 + 0), F(M2 + 1), nullptr, t64, 512, 64, 128, 1);
  gemm(h1, F(M2 + 4), F(M2 + 5), nullptr, scb, 512, 128, 128, 0);
  gemm(t64, F(M2 + 2), F(M2 + 3), scb, h2, 512, 128, 64, 0);
  // mlp3
  k_cat_fr<<<blk1(BATCH * 512 * 640), 256, 0, stream>>>(h2, out, cat640, BATCH * 512 * 640);
  gemm(cat640, F(M3 + 0), F(M3 + 1), nullptr, hbuf, 512, 128, 640, 1);
  gemm(cat640, F(M3 + 4), F(M3 + 5), nullptr, scb, 512, 128, 640, 0);
  gemm(hbuf, F(M3 + 2), F(M3 + 3), scb, h3, 512, 128, 128, 0);
  // head (output 2)
  gemm(h3, F(IN_4AW), F(IN_4AB), nullptr, t64, 512, 64, 128, 2);
  k_head<<<blk1(BATCH * 512 * 3), 256, 0, stream>>>(t64, F(IN_4BW), F(IN_4BB),
                                                    out + BATCH * 512, BATCH * 512 * 3);
}
